// LSTM_COMM_22222160789673
// MI455X (gfx1250) — matrix operand order repaired and checked by reading, not compiled, not run on hardware
//
#include <hip/hip_runtime.h>
#include <cstdint>
#include <cstddef>

// ---------------------------------------------------------------------------
// CDNA5 (gfx1250) bf16 WMMA pipeline for the 8-agent GRU comm network.
// All matmuls run on v_wmma_f32_16x16x32_bf16 (wave32, 16x16 tiles, f32 acc).
// ---------------------------------------------------------------------------

typedef __bf16 bf16_t;
typedef __attribute__((ext_vector_type(16))) __bf16 v16bf;
typedef __attribute__((ext_vector_type(8)))  __bf16 v8bf;
typedef __attribute__((ext_vector_type(8)))  float  v8f;

union V16U { v16bf v; v8bf h[2]; };

__device__ __forceinline__ v8f zero_v8f() {
  v8f z;
#pragma unroll
  for (int i = 0; i < 8; ++i) z[i] = 0.0f;
  return z;
}

// A fragment: 16x32 bf16 tile of row-major A[M,K].
// Layout used here (contiguous k, not ISA table order): lanes 0-15 -> row M=lane, K 0..15;
//             lanes 16-31 -> row M=lane-16, K 16..31.
// Row gather: memory row = (row0+r)*rowMul + rowAdd (for agent selection).
__device__ __forceinline__ v16bf load_a_frag(const bf16_t* __restrict__ A, int ld,
                                             int row0, int k0, int rowMul, int rowAdd,
                                             int lane) {
  int r  = lane & 15;
  int ks = (lane >> 4) << 4;  // 0 or 16
  const bf16_t* p = A + (size_t)((row0 + r) * rowMul + rowAdd) * ld + k0 + ks;
  V16U u;
  u.h[0] = *(const v8bf*)(p);        // K + {0..7}  or {16..23}
  u.h[1] = *(const v8bf*)(p + 8);    // K + {8..15} or {24..31}
  return u.v;
}

// B fragment: 32x16 bf16 tile of B = W^T where W[N,K] is row-major.
// Layout used here (contiguous k, not ISA table order): lanes 0-15 -> col N=lane, K 0..15; lanes 16-31 -> K 16..31.
// One contiguous 32-byte load per lane.
__device__ __forceinline__ v16bf load_b_frag(const bf16_t* __restrict__ W, int ld,
                                             int col0, int k0, int lane) {
  int n  = lane & 15;
  int ks = (lane >> 4) << 4;  // 0 or 16
  return *(const v16bf*)(W + (size_t)(col0 + n) * ld + k0 + ks);
}

__device__ __forceinline__ v8f wmma_bf16(v16bf a, v16bf b, v8f c) {
  // (neg_a, A, neg_b, B, c_mod, C, reuse_a, reuse_b)
  return __builtin_amdgcn_wmma_f32_16x16x32_bf16(false, a, false, b, (short)0, c,
                                                 false, false);
}

__device__ __forceinline__ float sigmoidf_fast(float x) {
  return 1.0f / (1.0f + __expf(-x));
}

// ---------------------------------------------------------------------------
// f32 -> bf16 conversion
// ---------------------------------------------------------------------------
__global__ __launch_bounds__(256)
void cvt_f32_bf16(const float* __restrict__ in, bf16_t* __restrict__ out, int n) {
  int i = blockIdx.x * blockDim.x + threadIdx.x;
  if (i < n) out[i] = (bf16_t)in[i];
}

// ---------------------------------------------------------------------------
// Generic GEMM: C[M,N] = act(A[M,K] @ W[N,K]^T + bias)
// Workgroup = 256 threads = 8 waves laid out 4(M) x 2(N); each wave owns a
// 32x32 output tile (2x2 WMMA 16x16 tiles) -> workgroup tile 128 x 64.
// Output optionally written as f32 (Cf) and/or bf16 (Cb).
// ---------------------------------------------------------------------------
__global__ __launch_bounds__(256)
void gemm_bf16_wmma(const bf16_t* __restrict__ A, const bf16_t* __restrict__ W,
                    const float* __restrict__ bias,
                    float* __restrict__ Cf, bf16_t* __restrict__ Cb,
                    int M, int N, int K, int act,
                    int rowMul, const int* __restrict__ rowAddP) {
  const int rowAdd = rowAddP ? rowAddP[0] : 0;
  const int lane = threadIdx.x & 31;
  const int wave = threadIdx.x >> 5;
  const int wm = wave & 3;
  const int wn = wave >> 2;
  const int row0 = blockIdx.y * 128 + wm * 32;
  const int col0 = blockIdx.x * 64  + wn * 32;

  v8f acc[2][2];
#pragma unroll
  for (int i = 0; i < 2; ++i)
#pragma unroll
    for (int j = 0; j < 2; ++j) acc[i][j] = zero_v8f();

  for (int k0 = 0; k0 < K; k0 += 32) {
    v16bf a0 = load_a_frag(A, K, row0 +  0, k0, rowMul, rowAdd, lane);
    v16bf a1 = load_a_frag(A, K, row0 + 16, k0, rowMul, rowAdd, lane);
    v16bf b0 = load_b_frag(W, K, col0 +  0, k0, lane);
    v16bf b1 = load_b_frag(W, K, col0 + 16, k0, lane);
    acc[0][0] = wmma_bf16(a0, b0, acc[0][0]);
    acc[0][1] = wmma_bf16(a0, b1, acc[0][1]);
    acc[1][0] = wmma_bf16(a1, b0, acc[1][0]);
    acc[1][1] = wmma_bf16(a1, b1, acc[1][1]);
  }

  // C/D layout: VGPR r -> row r + 8*(lane>=16); col = lane&15.
  const int cl = lane & 15;
  const int rh = (lane >> 4) * 8;
#pragma unroll
  for (int i = 0; i < 2; ++i) {
#pragma unroll
    for (int j = 0; j < 2; ++j) {
      const int col = col0 + j * 16 + cl;
      const float bv = bias ? bias[col] : 0.0f;
#pragma unroll
      for (int r = 0; r < 8; ++r) {
        const int row = row0 + i * 16 + rh + r;
        float v = acc[i][j][r] + bv;
        if (act == 1) v = v > 0.0f ? v : 0.0f;  // ReLU
        if (Cf) Cf[(size_t)row * N + col] = v;
        if (Cb) Cb[(size_t)row * N + col] = (bf16_t)v;
      }
    }
  }
}

// ---------------------------------------------------------------------------
// Fused GRU step: per output tile compute all six gate GEMMs
//   a_i{r,z,n} = c @ Wih_{r,z,n}^T ,  a_h{r,z,n} = h @ Whh_{r,z,n}^T
// then the elementwise GRU update in the epilogue; gi/gh never hit memory.
// useC == 0 handles the first step (c == 0  =>  gi = bih).
// Wave tile 32(M) x 16(N): each weight fragment feeds two WMMAs (12 WMMA /
// 14 loads per K-step). Workgroup tile 128(M, 4 waves) x 32(N, 2 waves).
// ---------------------------------------------------------------------------
__global__ __launch_bounds__(256)
void gru_step_wmma(const bf16_t* __restrict__ Cin,   // comm input (x of GRUCell)
                   const bf16_t* __restrict__ Hin,   // previous hidden
                   const bf16_t* __restrict__ Wih,   // [3H, H]
                   const bf16_t* __restrict__ Whh,   // [3H, H]
                   const float* __restrict__ bih,    // [3H]
                   const float* __restrict__ bhh,    // [3H]
                   bf16_t* __restrict__ Hout,        // [Mout, H] (compact)
                   int Mout, int H, int useC,
                   int rowMul, const int* __restrict__ rowAddP) {
  const int rowAdd = rowAddP ? rowAddP[0] : 0;
  const int lane = threadIdx.x & 31;
  const int wave = threadIdx.x >> 5;
  const int row0 = blockIdx.y * 128 + (wave & 3) * 32;  // two 16-row tiles
  const int col0 = blockIdx.x * 32 + (wave >> 2) * 16;

  v8f aH[2][3], aI[2][3];  // [row tile][gate r,z,n]
#pragma unroll
  for (int t = 0; t < 2; ++t)
#pragma unroll
    for (int g = 0; g < 3; ++g) { aH[t][g] = zero_v8f(); aI[t][g] = zero_v8f(); }

  for (int k0 = 0; k0 < H; k0 += 32) {
    v16bf h0 = load_a_frag(Hin, H, row0 +  0, k0, rowMul, rowAdd, lane);
    v16bf h1 = load_a_frag(Hin, H, row0 + 16, k0, rowMul, rowAdd, lane);
#pragma unroll
    for (int g = 0; g < 3; ++g) {
      v16bf w = load_b_frag(Whh, H, g * H + col0, k0, lane);
      aH[0][g] = wmma_bf16(h0, w, aH[0][g]);
      aH[1][g] = wmma_bf16(h1, w, aH[1][g]);
    }
    if (useC) {
      v16bf c0 = load_a_frag(Cin, H, row0 +  0, k0, rowMul, rowAdd, lane);
      v16bf c1 = load_a_frag(Cin, H, row0 + 16, k0, rowMul, rowAdd, lane);
#pragma unroll
      for (int g = 0; g < 3; ++g) {
        v16bf w = load_b_frag(Wih, H, g * H + col0, k0, lane);
        aI[0][g] = wmma_bf16(c0, w, aI[0][g]);
        aI[1][g] = wmma_bf16(c1, w, aI[1][g]);
      }
    }
  }

  const int cl = lane & 15;
  const int rh = (lane >> 4) * 8;
  const int col = col0 + cl;
  const float bir = bih[col], biz = bih[col + H], bin = bih[col + 2 * H];
  const float bhr = bhh[col], bhz = bhh[col + H], bhn = bhh[col + 2 * H];
#pragma unroll
  for (int t = 0; t < 2; ++t) {
#pragma unroll
    for (int r = 0; r < 8; ++r) {
      const int row = row0 + t * 16 + rh + r;
      const float gir = bir + (useC ? aI[t][0][r] : 0.0f);
      const float giz = biz + (useC ? aI[t][1][r] : 0.0f);
      const float gin = bin + (useC ? aI[t][2][r] : 0.0f);
      const float ghr = aH[t][0][r] + bhr;
      const float ghz = aH[t][1][r] + bhz;
      const float ghn = aH[t][2][r] + bhn;
      const float rg = sigmoidf_fast(gir + ghr);
      const float zg = sigmoidf_fast(giz + ghz);
      const float ng = tanhf(gin + rg * ghn);
      const float hold = (float)Hin[(size_t)(row * rowMul + rowAdd) * H + col];
      const float hnew = (1.0f - zg) * ng + zg * hold;
      Hout[(size_t)row * H + col] = (bf16_t)hnew;
    }
  }
}

// ---------------------------------------------------------------------------
// Agent communication: c[b,i,:] = (sum_j h[b,j,:] - h[b,i,:]) / nA
// One thread per (batch, channel); nA == 8.
// ---------------------------------------------------------------------------
__global__ __launch_bounds__(256)
void comm_kernel(const bf16_t* __restrict__ Hb, bf16_t* __restrict__ Cb,
                 int B, int nA, int H) {
  int idx = blockIdx.x * blockDim.x + threadIdx.x;
  if (idx >= B * H) return;
  int b = idx / H, col = idx - b * H;
  const bf16_t* p = Hb + (size_t)b * nA * H + col;
  float v[8];
  float s = 0.0f;
#pragma unroll
  for (int j = 0; j < 8; ++j) { v[j] = (float)p[(size_t)j * H]; s += v[j]; }
  const float inv = 1.0f / (float)nA;
  bf16_t* q = Cb + (size_t)b * nA * H + col;
#pragma unroll
  for (int j = 0; j < 8; ++j) q[(size_t)j * H] = (bf16_t)((s - v[j]) * inv);
}

// ---------------------------------------------------------------------------
// Host-side pipeline
// ---------------------------------------------------------------------------
extern "C" void kernel_launch(void* const* d_in, const int* in_sizes, int n_in,
                              void* d_out, int out_size, void* d_ws, size_t ws_size,
                              hipStream_t stream) {
  (void)in_sizes; (void)n_in; (void)out_size; (void)ws_size;
  const int B = 4096, NA = 8, D = 256, H = 512, O = 64;
  const int M = B * NA;  // 32768 rows

  const float* x   = (const float*)d_in[0];
  const float* W1  = (const float*)d_in[1];
  const float* b1  = (const float*)d_in[2];
  const float* W2  = (const float*)d_in[3];
  const float* b2  = (const float*)d_in[4];
  const float* Wih = (const float*)d_in[5];
  const float* Whh = (const float*)d_in[6];
  const float* bih = (const float*)d_in[7];
  const float* bhh = (const float*)d_in[8];
  const float* W3  = (const float*)d_in[9];
  const float* b3  = (const float*)d_in[10];
  const int* agent_ix = (const int*)d_in[11];  // read on device
  // d_in[12] is K == 3 in the reference setup; pipeline below is K = 3.
  float* out = (float*)d_out;

  char* ws = (char*)d_ws;
  size_t off = 0;
  auto alloc = [&](size_t bytes) -> void* {
    off = (off + 255) & ~(size_t)255;
    void* p = ws + off;
    off += bytes;
    return p;
  };
  bf16_t* xb   = (bf16_t*)alloc((size_t)M * D * 2);      // 16 MB
  bf16_t* w1b  = (bf16_t*)alloc((size_t)H * D * 2);
  bf16_t* w2b  = (bf16_t*)alloc((size_t)H * H * 2);
  bf16_t* wihb = (bf16_t*)alloc((size_t)3 * H * H * 2);
  bf16_t* whhb = (bf16_t*)alloc((size_t)3 * H * H * 2);
  bf16_t* w3b  = (bf16_t*)alloc((size_t)O * H * 2);
  bf16_t* t1   = (bf16_t*)alloc((size_t)M * H * 2);      // 32 MB
  bf16_t* hA   = (bf16_t*)alloc((size_t)M * H * 2);      // 32 MB
  bf16_t* hB   = (bf16_t*)alloc((size_t)M * H * 2);      // 32 MB
  bf16_t* cb   = (bf16_t*)alloc((size_t)M * H * 2);      // 32 MB
  bf16_t* hsel = (bf16_t*)alloc((size_t)B * H * 2);      //  4 MB

  auto cvt = [&](const float* src, bf16_t* dst, int n) {
    cvt_f32_bf16<<<(n + 255) / 256, 256, 0, stream>>>(src, dst, n);
  };
  cvt(x,   xb,   M * D);
  cvt(W1,  w1b,  H * D);
  cvt(W2,  w2b,  H * H);
  cvt(Wih, wihb, 3 * H * H);
  cvt(Whh, whhb, 3 * H * H);
  cvt(W3,  w3b,  O * H);

  // t1 = relu(x @ W1^T + b1)            [M, H]
  gemm_bf16_wmma<<<dim3(H / 64, M / 128), 256, 0, stream>>>(
      xb, w1b, b1, nullptr, t1, M, H, D, /*act=*/1, 1, nullptr);
  // h0 = t1 @ W2^T + b2                 -> hA
  gemm_bf16_wmma<<<dim3(H / 64, M / 128), 256, 0, stream>>>(
      t1, w2b, b2, nullptr, hA, M, H, H, 0, 1, nullptr);
  // GRU step 1 (c = 0 => gi = bih):     hA -> hB
  gru_step_wmma<<<dim3(H / 32, M / 128), 256, 0, stream>>>(
      nullptr, hA, wihb, whhb, bih, bhh, hB, M, H, /*useC=*/0, 1, nullptr);
  // c2 = comm(h1)
  comm_kernel<<<(B * H + 255) / 256, 256, 0, stream>>>(hB, cb, B, NA, H);
  // GRU step 2: (c2, hB) -> hA
  gru_step_wmma<<<dim3(H / 32, M / 128), 256, 0, stream>>>(
      cb, hB, wihb, whhb, bih, bhh, hA, M, H, 1, 1, nullptr);
  // c3 = comm(h2)
  comm_kernel<<<(B * H + 255) / 256, 256, 0, stream>>>(hA, cb, B, NA, H);
  // GRU step 3, only rows b*NA + agent_ix (4096 of 32768): -> hsel [B, H]
  gru_step_wmma<<<dim3(H / 32, B / 128), 256, 0, stream>>>(
      cb, hA, wihb, whhb, bih, bhh, hsel, B, H, 1, NA, agent_ix);
  // out = hsel @ W3^T + b3              [B, O] f32
  gemm_bf16_wmma<<<dim3(O / 64, B / 128), 256, 0, stream>>>(
      hsel, w3b, b3, out, nullptr, B, O, H, 0, 1, nullptr);
}
